// LookGround_38826504356675
// MI455X (gfx1250) — compile-verified, matrix operand order checked by reading, not run on hardware
//
#include <hip/hip_runtime.h>

#define B_ 16
#define C_ 256
#define H_ 48
#define W_ 160
#define KDIM 257
#define KPAD 288
#define NGROUPW (W_/16)

typedef __attribute__((ext_vector_type(16))) __bf16 v16bf;
typedef __attribute__((ext_vector_type(8)))  float v8f;

union BfVec { unsigned int u[8]; v16bf v; };

__device__ __forceinline__ unsigned short f2bf(float f) {
    unsigned int u = __float_as_uint(f);
    u += 0x7FFFu + ((u >> 16) & 1u);   // round-to-nearest-even
    return (unsigned short)(u >> 16);
}

// ---------------------------------------------------------------------------
// Pass 0: convert w_ext (256 x 257 fp32) -> K-padded bf16 matrix (256 x 288)
// ---------------------------------------------------------------------------
__global__ void __launch_bounds__(256) prep_wext(const float* __restrict__ w_ext,
                                                 unsigned short* __restrict__ wA) {
    int i = blockIdx.x * 256 + threadIdx.x;      // i < 256*KPAD exactly
    int o = i / KPAD, k = i % KPAD;
    wA[i] = (k < KDIM) ? f2bf(w_ext[o * KDIM + k]) : (unsigned short)0;
}

// ---------------------------------------------------------------------------
// Pass 1: 3x3 conv (256ch -> 1ch) + tanh -> disp buffer (B,H,W)
// ---------------------------------------------------------------------------
__global__ void __launch_bounds__(256) disp_conv(const float* __restrict__ x,
                                                 const float* __restrict__ w_disp,
                                                 const float* __restrict__ b_disp,
                                                 float* __restrict__ dispOut) {
    __shared__ float sW[C_ * 9];
    for (int i = threadIdx.x; i < C_ * 9; i += 256) sW[i] = w_disp[i];
    __syncthreads();

    int idx = blockIdx.x * 256 + threadIdx.x;    // < B*H*W
    int w = idx % W_;
    int h = (idx / W_) % H_;
    int b = idx / (W_ * H_);

    float s = 0.f;
    const float* xb = x + (size_t)b * C_ * H_ * W_;
    for (int c = 0; c < C_; ++c) {
        const float* xc = xb + (size_t)c * H_ * W_;
        const float* wc = &sW[c * 9];
        #pragma unroll
        for (int dy = 0; dy < 3; ++dy) {
            int y = h + dy - 1;
            if (y < 0 || y >= H_) continue;      // SAME zero padding
            const float* row = xc + y * W_;
            if (w > 0)       s += row[w - 1] * wc[dy * 3 + 0];
                             s += row[w    ] * wc[dy * 3 + 1];
            if (w < W_ - 1)  s += row[w + 1] * wc[dy * 3 + 2];
        }
    }
    dispOut[idx] = 0.1f * tanhf(s + b_disp[0]);
}

// ---------------------------------------------------------------------------
// Pass 2: y-lerp gather -> LDS bf16, 257-K GEMM via v_wmma_f32_16x16x32_bf16,
//         fused bias + alpha + residual + relu epilogue.
// Block = 256 threads (8 waves); block handles 16 pixels x all 256 out chans.
// ---------------------------------------------------------------------------
__global__ void __launch_bounds__(256) lookground_wmma(
        const float* __restrict__ x, const float* __restrict__ P2,
        const unsigned short* __restrict__ wA, const float* __restrict__ b_ext,
        const float* __restrict__ alpha, const float* __restrict__ disp,
        float* __restrict__ out) {
    __shared__ unsigned short sB[16 * KPAD];     // sampled vectors, bf16
    __shared__ float sWy[16];
    __shared__ int   sY0[16], sY1[16];

    const int g  = blockIdx.x;
    const int wg = g % NGROUPW;
    const int h  = (g / NGROUPW) % H_;
    const int b  = g / (NGROUPW * H_);
    const int w0 = wg * 16;
    const int t  = threadIdx.x;

    // ---- per-pixel warp parameters (threads 0..15) ----
    if (t < 16) {
        const int p = t, w = w0 + p;
        const float fy = P2[b * 12 + 5] * (1.f / 16.f);
        const float cy = P2[b * 12 + 6] * (1.f / 16.f);
        const float Ty = P2[b * 12 + 7] * (1.f / 16.f);
        const float inv_dd = 1.f / (fabsf(fy * 1.65f + Ty) + 1e-10f);
        const float ysb = fmaxf(0.f, 1.535f * ((float)h - cy) /
                                     (2.f * (1.65f - 0.5f * 1.535f))) * (1.f / (0.5f * H_));
        const float ybase = -1.f + 2.f * (float)h / (float)(H_ - 1);
        const float gy = ybase + ysb + disp[((size_t)b * H_ + h) * W_ + w];
        float iy = (gy + 1.f) * 0.5f * (float)(H_ - 1);
        iy = fminf(fmaxf(iy, 0.f), (float)(H_ - 1));
        const float f0 = floorf(iy);
        const int y0 = (int)f0;
        const int y1 = min(y0 + 1, H_ - 1);
        const float wy = iy - f0;
        // disparity pseudo-channel sampled at y0/y1
        const float d0 = fmaxf(0.f, fy * 0.54f * ((float)y0 - cy) * inv_dd);
        const float d1 = fmaxf(0.f, fy * 0.54f * ((float)y1 - cy) * inv_dd);
        sY0[p] = y0; sY1[p] = y1; sWy[p] = wy;
        sB[p * KPAD + 0] = f2bf(d0 + wy * (d1 - d0));
    }
    // zero the K padding (257..287)
    {
        int p = t & 15;
        int kk = 257 + (t >> 4);
        if (kk < KPAD) sB[p * KPAD + kk] = 0;
        kk += 16;
        if (kk < KPAD) sB[p * KPAD + kk] = 0;
    }
    __syncthreads();

    // ---- cooperative y-lerp gather of 256 channels -> bf16 in LDS ----
    {
        const int p  = t & 15;
        const int kl = 1 + (t >> 4);
        const int y0 = sY0[p], y1 = sY1[p];
        const float wy = sWy[p];
        const size_t base0 = ((size_t)(b * C_) * H_ + y0) * W_ + w0 + p;
        const size_t base1 = ((size_t)(b * C_) * H_ + y1) * W_ + w0 + p;
        #pragma unroll 4
        for (int j = 0; j < 16; ++j) {
            int k = kl + 16 * j;                 // covers 1..256
            size_t coff = (size_t)(k - 1) * H_ * W_;
            float v0 = x[base0 + coff];
            float v1 = x[base1 + coff];
            sB[p * KPAD + k] = f2bf(v0 + wy * (v1 - v0));
        }
    }
    __syncthreads();

    // ---- WMMA: each wave owns two 16(M) x 16(N) tiles, shared B fragment ----
    const int lane = t & 31;
    const int wv   = t >> 5;                     // 0..7
    const int half = lane >> 4;                  // 0..1
    const int mrow = lane & 15;
    const int mt0 = wv * 2, mt1 = wv * 2 + 1;

    const unsigned short* a0 = wA + (size_t)(mt0 * 16 + mrow) * KPAD;
    const unsigned short* a1 = wA + (size_t)(mt1 * 16 + mrow) * KPAD;
    const unsigned short* brow = &sB[(lane & 15) * KPAD];

    const int p = lane & 15;
    const size_t pixBase = ((size_t)b * C_ * H_ + h) * W_ + w0 + p;
    __builtin_prefetch(&x[pixBase + (size_t)(mt0 * 16) * H_ * W_], 0, 0);
    __builtin_prefetch(&x[pixBase + (size_t)(mt1 * 16) * H_ * W_], 0, 0);

    v8f acc0 = {}; v8f acc1 = {};
    for (int kb = 0; kb < KPAD; kb += 32) {
        BfVec A0, A1, Bv;
        #pragma unroll
        for (int v = 0; v < 8; ++v) {
            // 16-bit A/B fragment layout (ISA 7.12.2): VGPR v holds packed
            // pair K = 2*(v&3) + 8*half + 16*(v>>2), (K, K+1)
            int k = kb + ((v & 3) << 1) + (half << 3) + ((v >> 2) << 4);
            A0.u[v] = *reinterpret_cast<const unsigned int*>(a0 + k);
            A1.u[v] = *reinterpret_cast<const unsigned int*>(a1 + k);
            Bv.u[v] = *reinterpret_cast<const unsigned int*>(brow + k);
        }
        acc0 = __builtin_amdgcn_wmma_f32_16x16x32_bf16(
                   false, A0.v, false, Bv.v, (short)0, acc0, false, false);
        acc1 = __builtin_amdgcn_wmma_f32_16x16x32_bf16(
                   false, A1.v, false, Bv.v, (short)0, acc1, false, false);
    }

    // ---- fused epilogue: relu(x + (gemm + b_ext) * alpha) ----
    const float a_ = alpha[0];
    #pragma unroll
    for (int r = 0; r < 8; ++r) {
        int o0 = mt0 * 16 + (half << 3) + r;     // C/D layout: M = r + 8*half
        int o1 = mt1 * 16 + (half << 3) + r;
        size_t i0 = pixBase + (size_t)o0 * H_ * W_;
        size_t i1 = pixBase + (size_t)o1 * H_ * W_;
        out[i0] = fmaxf(0.f, x[i0] + (acc0[r] + b_ext[o0]) * a_);
        out[i1] = fmaxf(0.f, x[i1] + (acc1[r] + b_ext[o1]) * a_);
    }
}

// ---------------------------------------------------------------------------
extern "C" void kernel_launch(void* const* d_in, const int* in_sizes, int n_in,
                              void* d_out, int out_size, void* d_ws, size_t ws_size,
                              hipStream_t stream) {
    const float* feats  = (const float*)d_in[0];
    const float* P2     = (const float*)d_in[1];
    const float* w_disp = (const float*)d_in[2];
    const float* b_disp = (const float*)d_in[3];
    const float* w_ext  = (const float*)d_in[4];
    const float* b_ext  = (const float*)d_in[5];
    const float* alpha  = (const float*)d_in[6];
    float* out = (float*)d_out;

    float* dispBuf = (float*)d_ws;                                   // 0.47 MB
    unsigned short* wA = (unsigned short*)((char*)d_ws + 524288);    // 147 KB

    prep_wext<<<(256 * KPAD) / 256, 256, 0, stream>>>(w_ext, wA);
    disp_conv<<<(B_ * H_ * W_) / 256, 256, 0, stream>>>(feats, w_disp, b_disp, dispBuf);
    lookground_wmma<<<B_ * H_ * NGROUPW, 256, 0, stream>>>(
        feats, P2, wA, b_ext, alpha, dispBuf, out);
}
